// guided_CAM_Module2_50646254355128
// MI455X (gfx1250) — matrix operand order repaired and checked by reading, not compiled, not run on hardware
//
#include <hip/hip_runtime.h>
#include <hip/hip_bf16.h>
#include <cstdint>
#include <cstddef>

typedef __bf16 bf16;
typedef __attribute__((ext_vector_type(16))) bf16  bf16x16;
typedef __attribute__((ext_vector_type(8)))  bf16  bf16x8;
typedef __attribute__((ext_vector_type(8)))  float f32x8;

#define B_   8
#define C_   512
#define H_   96
#define W_   96
#define HW_  9216

static __device__ __forceinline__ bf16x16 join8(bf16x8 lo, bf16x8 hi) {
  return __builtin_shufflevector(lo, hi, 0,1,2,3,4,5,6,7,8,9,10,11,12,13,14,15);
}

// ---------------------------------------------------------------------------
// Fold BN scale into weights (fp32 -> bf16) and compute fused bias.
//   y = ReLU((W x - mean)*inv + bias) == ReLU((W*inv) x + (bias - mean*inv))
// ---------------------------------------------------------------------------
__global__ __launch_bounds__(256) void prep_w_kernel(
    const float* __restrict__ w, const float* __restrict__ scale,
    const float* __restrict__ bias, const float* __restrict__ mean,
    const float* __restrict__ var, bf16* __restrict__ wb,
    float* __restrict__ bp) {
  int i = blockIdx.x * 256 + threadIdx.x;
  if (i >= C_ * C_) return;
  int o = i >> 9;
  float inv = scale[o] * rsqrtf(var[o] + 1e-5f);
  wb[i] = (bf16)(w[i] * inv);
  if ((i & (C_ - 1)) == 0) bp[o] = bias[o] - mean[o] * inv;
}

// ---------------------------------------------------------------------------
// LDS-tiled transpose: f32 [z][R][N] -> bf16 [z][N][R]   (K-contiguous for WMMA B)
// block (32,8), grid (N/32, R/32, z)
// ---------------------------------------------------------------------------
__global__ __launch_bounds__(256) void transpose_f32_to_bf16T(
    const float* __restrict__ in, bf16* __restrict__ out, int R, int N) {
  __shared__ float tile[32][33];
  const float* inb = in + (size_t)blockIdx.z * R * N;
  bf16* outb = out + (size_t)blockIdx.z * R * N;
  int r0 = blockIdx.y * 32, c0 = blockIdx.x * 32;
  int tx = threadIdx.x, ty = threadIdx.y;
#pragma unroll
  for (int i = 0; i < 32; i += 8)
    tile[ty + i][tx] = inb[(size_t)(r0 + ty + i) * N + (c0 + tx)];
  __syncthreads();
#pragma unroll
  for (int i = 0; i < 32; i += 8)
    outb[(size_t)(c0 + ty + i) * R + (r0 + tx)] = (bf16)tile[tx][ty + i];
}

// ---------------------------------------------------------------------------
// WMMA bf16 GEMM: D[b][m][n] = sum_k A[m][k] * BT[b][n][k]   (M=K=512, N=9216)
// MODE 0: ReLU(acc+bias) -> bf16 map      MODE 1: ReLU(acc+bias) -> f32 map
// MODE 2: gamma*acc + res -> f32 out      (A batched via aBatchStride)
// 256 threads = 8 waves; wave tile 32x32 (2x2 WMMA); block tile 128(M) x 64(N)
// ---------------------------------------------------------------------------
template <int MODE>
__global__ __launch_bounds__(256) void gemm_wmma_kernel(
    const bf16* __restrict__ A, size_t aBatchStride,
    const bf16* __restrict__ BT, const float* __restrict__ bias,
    const float* __restrict__ res, const float* __restrict__ gamma,
    bf16* __restrict__ outb, float* __restrict__ outf) {
  constexpr int K = C_;
  const int b = blockIdx.z;
  const bf16* Ab = A + (size_t)b * aBatchStride;
  const bf16* Bb = BT + (size_t)b * HW_ * C_;
  const int lane = threadIdx.x & 31;
  const int wid = threadIdx.x >> 5;
  const int mWave = blockIdx.y * 128 + (wid >> 1) * 32;
  const int nWave = blockIdx.x * 64 + (wid & 1) * 32;
  const int l15 = lane & 15;
  const int hi = lane >> 4;

  // A frag (16x32 bf16): lane<16 -> K 0..15; lane>=16 -> K 16..31
  const bf16* a0p = Ab + (size_t)(mWave + l15) * K + hi * 16;
  const bf16* a1p = a0p + (size_t)16 * K;
  // B frag (32x16 bf16, BT is [n][k]): lane<16 -> K 0..15; lane>=16 -> K 16..31
  const bf16* b0p = Bb + (size_t)(nWave + l15) * K + hi * 16;
  const bf16* b1p = b0p + (size_t)16 * K;

  f32x8 acc00 = {0.f,0.f,0.f,0.f,0.f,0.f,0.f,0.f};
  f32x8 acc01 = acc00, acc10 = acc00, acc11 = acc00;

  for (int k = 0; k < K; k += 32) {
    bf16x16 a0 = *(const bf16x16*)(a0p + k);
    bf16x16 a1 = *(const bf16x16*)(a1p + k);
    bf16x16 b0 = *(const bf16x16*)(b0p + k);
    bf16x16 b1 = *(const bf16x16*)(b1p + k);
    acc00 = __builtin_amdgcn_wmma_f32_16x16x32_bf16(false, a0, false, b0, (short)0, acc00, false, false);
    acc01 = __builtin_amdgcn_wmma_f32_16x16x32_bf16(false, a0, false, b1, (short)0, acc01, false, false);
    acc10 = __builtin_amdgcn_wmma_f32_16x16x32_bf16(false, a1, false, b0, (short)0, acc10, false, false);
    acc11 = __builtin_amdgcn_wmma_f32_16x16x32_bf16(false, a1, false, b1, (short)0, acc11, false, false);
  }

  const float g = (MODE == 2) ? gamma[0] : 0.0f;
  auto store8 = [&](const f32x8& acc, int mt, int nt) {
#pragma unroll
    for (int r = 0; r < 8; ++r) {
      // C/D layout: VGPR r, lanes 0-15 -> M=r, lanes 16-31 -> M=8+r, N=lane&15
      int m = mWave + mt * 16 + hi * 8 + r;
      int n = nWave + nt * 16 + l15;
      size_t idx = ((size_t)b * C_ + m) * HW_ + n;
      float v = acc[r];
      if (MODE == 0) {
        v = fmaxf(v + bias[m], 0.0f);
        outb[idx] = (bf16)v;
      } else if (MODE == 1) {
        v = fmaxf(v + bias[m], 0.0f);
        outf[idx] = v;
      } else {
        outf[idx] = fmaf(g, v, res[idx]);
      }
    }
  };
  store8(acc00, 0, 0);
  store8(acc01, 0, 1);
  store8(acc10, 1, 0);
  store8(acc11, 1, 1);
}

// ---------------------------------------------------------------------------
// Pyramid pooling: one block per (b,channel) 96x96 bf16 map -> 50 fp32 cells
// cells: [0]=1x1, [1..4]=2x2, [5..13]=3x3, [14..49]=6x6  (row-major h*s+w)
// ---------------------------------------------------------------------------
__global__ __launch_bounds__(256) void pool_kernel(
    const bf16* __restrict__ maps, float* __restrict__ pools) {
  const bf16* m = maps + (size_t)blockIdx.x * HW_;
  __shared__ float cells[50];
  if (threadIdx.x < 50) cells[threadIdx.x] = 0.0f;
  __syncthreads();
  float s1 = 0.0f;
  for (int i = threadIdx.x; i < HW_; i += 256) {
    float v = (float)m[i];
    int h = i / W_, w = i - h * W_;
    s1 += v;
    atomicAdd(&cells[1 + (h / 48) * 2 + (w / 48)], v);
    atomicAdd(&cells[5 + (h / 32) * 3 + (w / 32)], v);
    atomicAdd(&cells[14 + (h / 16) * 6 + (w / 16)], v);
  }
  atomicAdd(&cells[0], s1);
  __syncthreads();
  if (threadIdx.x < 50) {
    int j = threadIdx.x;
    float inv = (j < 1) ? (1.0f / 9216.0f)
              : (j < 5) ? (1.0f / 2304.0f)
              : (j < 14) ? (1.0f / 1024.0f)
                         : (1.0f / 256.0f);
    pools[(size_t)blockIdx.x * 50 + j] = cells[j] * inv;
  }
}

// ---------------------------------------------------------------------------
// energy[b][c][d] = sum_n q[b][c][n] * k[b][d][n]   (n=50, tiny GEMM)
// grid (C/256, C, B), 256 threads
// ---------------------------------------------------------------------------
__global__ __launch_bounds__(256) void energy_kernel(
    const float* __restrict__ qp, const float* __restrict__ kp,
    float* __restrict__ energy) {
  int b = blockIdx.z, c = blockIdx.y;
  int d = blockIdx.x * 256 + threadIdx.x;
  __shared__ float qrow[50];
  if (threadIdx.x < 50) qrow[threadIdx.x] = qp[((size_t)b * C_ + c) * 50 + threadIdx.x];
  __syncthreads();
  const float* krow = kp + ((size_t)b * C_ + d) * 50;
  float s = 0.0f;
#pragma unroll
  for (int n = 0; n < 50; ++n) s = fmaf(qrow[n], krow[n], s);
  energy[((size_t)b * C_ + c) * C_ + d] = s;
}

// ---------------------------------------------------------------------------
// softmax(rowmax - e) over last dim == exp(rowmin - e)/sum(exp(rowmin - e))
// one 256-thread block per row of 512; output bf16 (K-contiguous for WMMA A)
// ---------------------------------------------------------------------------
__global__ __launch_bounds__(256) void softmax_kernel(
    const float* __restrict__ energy, bf16* __restrict__ att) {
  size_t row = (size_t)blockIdx.y * C_ + blockIdx.x;
  const float* e = energy + row * C_;
  int tid = threadIdx.x;
  float v0 = e[tid], v1 = e[tid + 256];
  __shared__ float red[256];
  red[tid] = fminf(v0, v1);
  __syncthreads();
  for (int s = 128; s > 0; s >>= 1) {
    if (tid < s) red[tid] = fminf(red[tid], red[tid + s]);
    __syncthreads();
  }
  float mn = red[0];
  __syncthreads();
  float p0 = __expf(mn - v0), p1 = __expf(mn - v1);
  red[tid] = p0 + p1;
  __syncthreads();
  for (int s = 128; s > 0; s >>= 1) {
    if (tid < s) red[tid] += red[tid + s];
    __syncthreads();
  }
  float inv = 1.0f / red[0];
  att[row * C_ + tid] = (bf16)(p0 * inv);
  att[row * C_ + tid + 256] = (bf16)(p1 * inv);
}

// ---------------------------------------------------------------------------
extern "C" void kernel_launch(void* const* d_in, const int* in_sizes, int n_in,
                              void* d_out, int out_size, void* d_ws, size_t ws_size,
                              hipStream_t stream) {
  const float* x = (const float*)d_in[0];
  const float* wq = (const float*)d_in[1];
  const float* gamma = (const float*)d_in[16];
  const float* wk = (const float*)d_in[6];
  const float* wv = (const float*)d_in[11];
  float* out = (float*)d_out;

  uint8_t* ws = (uint8_t*)d_ws;
  size_t off = 0;
  auto alloc = [&](size_t bytes) -> void* {
    size_t cur = (off + 255) & ~(size_t)255;
    off = cur + bytes;
    return (void*)(ws + cur);
  };

  const size_t mapElems = (size_t)B_ * C_ * HW_;
  bf16* xT = (bf16*)alloc(mapElems * 2);              // x transposed [b][hw][c] bf16
  bf16* wqB = (bf16*)alloc((size_t)C_ * C_ * 2);
  bf16* wkB = (bf16*)alloc((size_t)C_ * C_ * 2);
  bf16* wvB = (bf16*)alloc((size_t)C_ * C_ * 2);
  float* bq = (float*)alloc(C_ * 4);
  float* bk = (float*)alloc(C_ * 4);
  float* bv = (float*)alloc(C_ * 4);
  bf16* qMap = (bf16*)alloc(mapElems * 2);            // [b][o][hw] bf16
  bf16* kMap = (bf16*)alloc(mapElems * 2);
  float* vF = (float*)alloc(mapElems * 4);            // v map fp32 (residual)
  bf16* vT = (bf16*)alloc(mapElems * 2);              // v transposed [b][hw][o] bf16
  float* qPool = (float*)alloc((size_t)B_ * C_ * 50 * 4);
  float* kPool = (float*)alloc((size_t)B_ * C_ * 50 * 4);
  float* energyBuf = (float*)alloc((size_t)B_ * C_ * C_ * 4);
  bf16* att = (bf16*)alloc((size_t)B_ * C_ * C_ * 2);
  (void)ws_size; (void)n_in; (void)in_sizes; (void)out_size;

  // 1) fold BN into weights
  dim3 gPrep((C_ * C_ + 255) / 256);
  prep_w_kernel<<<gPrep, 256, 0, stream>>>(wq, (const float*)d_in[2], (const float*)d_in[3],
                                           (const float*)d_in[4], (const float*)d_in[5], wqB, bq);
  prep_w_kernel<<<gPrep, 256, 0, stream>>>(wk, (const float*)d_in[7], (const float*)d_in[8],
                                           (const float*)d_in[9], (const float*)d_in[10], wkB, bk);
  prep_w_kernel<<<gPrep, 256, 0, stream>>>(wv, (const float*)d_in[12], (const float*)d_in[13],
                                           (const float*)d_in[14], (const float*)d_in[15], wvB, bv);

  // 2) x -> bf16, transposed to [b][hw][c]
  dim3 tBlk(32, 8);
  dim3 tGrd(HW_ / 32, C_ / 32, B_);
  transpose_f32_to_bf16T<<<tGrd, tBlk, 0, stream>>>(x, xT, C_, HW_);

  // 3) conv+BN+ReLU GEMMs (WMMA bf16)
  dim3 gGrd(HW_ / 64, C_ / 128, B_);
  gemm_wmma_kernel<0><<<gGrd, 256, 0, stream>>>(wqB, 0, xT, bq, vF, gamma, qMap, vF);
  gemm_wmma_kernel<0><<<gGrd, 256, 0, stream>>>(wkB, 0, xT, bk, vF, gamma, kMap, vF);
  gemm_wmma_kernel<1><<<gGrd, 256, 0, stream>>>(wvB, 0, xT, bv, vF, gamma, qMap, vF);

  // 4) v -> bf16 transposed [b][hw][o] for attention GEMM
  transpose_f32_to_bf16T<<<tGrd, tBlk, 0, stream>>>(vF, vT, C_, HW_);

  // 5) pyramid pooling of q,k
  pool_kernel<<<B_ * C_, 256, 0, stream>>>(qMap, qPool);
  pool_kernel<<<B_ * C_, 256, 0, stream>>>(kMap, kPool);

  // 6) energy = q . k^T over 50 pooled cols
  dim3 eGrd(C_ / 256, C_, B_);
  energy_kernel<<<eGrd, 256, 0, stream>>>(qPool, kPool, energyBuf);

  // 7) softmax(max - e) -> attention bf16
  dim3 sGrd(C_, B_);
  softmax_kernel<<<sGrd, 256, 0, stream>>>(energyBuf, att);

  // 8) out = gamma * (att @ v) + v   (WMMA bf16, fp32 residual epilogue)
  gemm_wmma_kernel<2><<<gGrd, 256, 0, stream>>>(att, (size_t)C_ * C_, vT, bq, vF, gamma, qMap, out);
}
